// ProposalLayer_31705448579231
// MI455X (gfx1250) — matrix operand order repaired and checked by reading, not compiled, not run on hardware
//
#include <hip/hip_runtime.h>
#include <hip/hip_bf16.h>

#define BATCH    16
#define NBOX     200000
#define NT       4096      // pre-NMS top-k
#define POST     2000      // final outputs per batch
#define IOU_CUT  0.7f
#define NTHREADS 1024
#define EQ_CAP   2048

typedef __attribute__((ext_vector_type(16))) _Float16 v16h;
typedef __attribute__((ext_vector_type(8)))  float    v8f;

// Order-preserving uint transform for f32 (monotonic for all floats).
__device__ __forceinline__ unsigned int fkey(float f) {
  unsigned int u = __float_as_uint(f);
  return (u & 0x80000000u) ? ~u : (u | 0x80000000u);
}
__device__ __forceinline__ float fdec(unsigned int e) {
  return __uint_as_float((e & 0x80000000u) ? (e ^ 0x80000000u) : ~e);
}

// 32-element inclusive prefix sum across one wave32 using two WMMA f16 ops.
// v[0..31] must be exactly representable in f16 (counts <= 2048 here).
// A[i][k] = v[k] for all rows i (16x32).  Layout used here, contiguous k (16-bit A 16x32):
//   lanes 0-15 : elem e -> K=e
//   lanes 16-31: elem e -> K=16+e
// B1[k][j] = (k <= j), B2[k][j] = (k <= j+16).  Same contiguous k layout (16-bit B 32x16):
//   lanes 0-15 : col j=lane,    elem e -> K=e
//   lanes 16-31: col j=lane-16, elem e -> K=16+e
// D (16x16 f32): lane l, vgpr r -> D[M][N], N=l%16. Rows identical, so d[0]
// of WMMA1 gives prefix_{l} for l<16 and d[0] of WMMA2 gives prefix_{l} for l>=16.
__device__ __forceinline__ float wave_scan32_wmma(int lane, const float* v) {
  v16h a = {}, b1 = {}, b2 = {};
  const int k0 = (lane < 16) ? 0 : 16;
  const int k1 = k0 + 8;
#pragma unroll
  for (int e = 0; e < 8; ++e) {
    a[e]     = (_Float16)v[k0 + e];
    a[8 + e] = (_Float16)v[k1 + e];
  }
  const int j = lane & 15;
#pragma unroll
  for (int e = 0; e < 16; ++e) {
    const int k = (lane < 16) ? e : (16 + e);
    b1[e] = (_Float16)((k <= j)      ? 1.0f : 0.0f);
    b2[e] = (_Float16)((k <= j + 16) ? 1.0f : 0.0f);
  }
  v8f c = {};
  v8f d1 = __builtin_amdgcn_wmma_f32_16x16x32_f16(false, a, false, b1, (short)0, c, false, false);
  v8f d2 = __builtin_amdgcn_wmma_f32_16x16x32_f16(false, a, false, b2, (short)0, c, false, false);
  return (lane < 16) ? d1[0] : d2[0];
}

__global__ __launch_bounds__(NTHREADS)
void proposal_nms_kernel(const float* __restrict__ boxes,
                         const float* __restrict__ scores,
                         float* __restrict__ outB,
                         float* __restrict__ outS) {
  __shared__ unsigned long long s_keys[NT];   // 32 KB: (score_key<<32)|~idx
  __shared__ unsigned int s_hist[4096];       // 16 KB: radix histograms
  __shared__ unsigned int s_eq[EQ_CAP];       // 8 KB : indices equal to threshold
  __shared__ unsigned int s_gsum[64];
  __shared__ float        s_stage[5 * 32];    // chunk pivot boxes: y1,x1,y2,x2,area
  __shared__ unsigned int s_nib[8];
  __shared__ unsigned int s_surv;
  __shared__ unsigned int s_ctrl[8];          // [1]=need/rem [2]=chosen [3]=eqCnt [4]=gatherCnt
  __shared__ float        s_scan[NTHREADS];   // 4 KB : per-thread keep counts
  __shared__ float        s_wsum[64];         // wave sums + exclusive offsets

  const int b    = blockIdx.x;
  const int tid  = threadIdx.x;
  const int lane = tid & 31;
  const int wave = tid >> 5;
  const float* sp = scores + (size_t)b * NBOX;
  const float* bp = boxes  + (size_t)b * NBOX * 4;

  // ---- zero this batch's output region (harness poisons d_out) ----
  for (int i = tid; i < POST * 4; i += NTHREADS) outB[(size_t)b * POST * 4 + i] = 0.0f;
  for (int i = tid; i < POST;     i += NTHREADS) outS[(size_t)b * POST + i]     = 0.0f;

  if (tid == 0) s_ctrl[1] = NT;

  // ================= Phase A: radix select the top-NT threshold =============
  unsigned int pref = 0, prefMask = 0;
  const int shifts[3] = {20, 8, 0};
  const int nbits[3]  = {12, 12, 8};
#pragma unroll 1
  for (int pass = 0; pass < 3; ++pass) {
    const int shift = shifts[pass];
    const int bins  = 1 << nbits[pass];
    for (int i = tid; i < bins; i += NTHREADS) s_hist[i] = 0u;
    __syncthreads();
    for (int i = tid; i < NBOX; i += NTHREADS) {
      __builtin_prefetch(sp + i + NTHREADS * 8, 0, 1);   // global_prefetch_b8
      const unsigned int u = fkey(sp[i]);
      if ((u & prefMask) == pref)
        atomicAdd(&s_hist[(u >> shift) & (unsigned)(bins - 1)], 1u);
    }
    __syncthreads();
    const int G = bins >> 6;              // groups of 64 bins
    if (tid < G) {
      unsigned int ssum = 0;
      for (int q = 0; q < 64; ++q) ssum += s_hist[tid * 64 + q];
      s_gsum[tid] = ssum;
    }
    __syncthreads();
    if (tid == 0) {
      unsigned int need = s_ctrl[1];
      unsigned int running = 0;
      int g = G - 1;
      for (; g > 0; --g) {
        const unsigned int h = s_gsum[g];
        if (running + h >= need) break;
        running += h;
      }
      int chosen = g * 64;
      for (int d = g * 64 + 63; d >= g * 64; --d) {
        const unsigned int h = s_hist[d];
        if (running + h >= need) { chosen = d; break; }
        running += h;
      }
      s_ctrl[1] = need - running;         // still needed from the == bin
      s_ctrl[2] = (unsigned int)chosen;
    }
    __syncthreads();
    pref     |= s_ctrl[2] << shift;
    prefMask |= (unsigned int)(bins - 1) << shift;
  }
  const unsigned int T = pref;            // exact 32-bit threshold key

  // ---- resolve ties at the threshold by lowest index (jax top_k order) ----
  if (tid == 0) { s_ctrl[3] = 0; s_ctrl[4] = 0; }
  __syncthreads();
  for (int i = tid; i < NBOX; i += NTHREADS) {
    if (fkey(sp[i]) == T) {
      const unsigned int p = atomicAdd(&s_ctrl[3], 1u);
      if (p < EQ_CAP) s_eq[p] = (unsigned int)i;
    }
  }
  __syncthreads();
  const unsigned int rem = s_ctrl[1];
  unsigned int eqCount = s_ctrl[3];
  if (eqCount > EQ_CAP) eqCount = EQ_CAP;

  // ================= Phase B: gather the NT selected (score,idx) keys =======
  for (int i = tid; i < NBOX; i += NTHREADS) {
    __builtin_prefetch(sp + i + NTHREADS * 8, 0, 1);
    const unsigned int u = fkey(sp[i]);
    bool inc = (u > T);
    if (!inc && u == T) {
      if (s_ctrl[3] <= rem) {
        inc = true;                       // all equals fit
      } else {
        unsigned int csm = 0;
        for (unsigned int e = 0; e < eqCount; ++e)
          csm += (s_eq[e] < (unsigned int)i) ? 1u : 0u;
        inc = (csm < rem);                // keep the `rem` lowest indices
      }
    }
    if (inc) {
      const unsigned int p = atomicAdd(&s_ctrl[4], 1u);
      if (p < NT)
        s_keys[p] = ((unsigned long long)u << 32) | (unsigned int)~(unsigned int)i;
    }
  }
  __syncthreads();
  unsigned int got = s_ctrl[4]; if (got > NT) got = NT;
  for (int p = (int)got + tid; p < NT; p += NTHREADS) s_keys[p] = 0ull;
  __syncthreads();

  // ================= Phase C: bitonic sort NT keys, descending ==============
  for (unsigned int k = 2; k <= NT; k <<= 1) {
    for (unsigned int jj = k >> 1; jj > 0; jj >>= 1) {
      for (unsigned int x = tid; x < NT; x += NTHREADS) {
        const unsigned int ix = x ^ jj;
        if (ix > x) {
          const unsigned long long va = s_keys[x], vb = s_keys[ix];
          const bool descSeg = ((x & k) == 0);
          if (descSeg ? (va < vb) : (va > vb)) { s_keys[x] = vb; s_keys[ix] = va; }
        }
      }
      __syncthreads();
    }
  }

  // ================= Phase D: load the 4 boxes this thread owns =============
  float ry1[4], rx1[4], ry2[4], rx2[4], rar[4], rsc[4];
  unsigned int keepm = 0xFu;
#pragma unroll
  for (int m = 0; m < 4; ++m) {
    const int slot = tid * 4 + m;
    const unsigned long long key = s_keys[slot];
    const unsigned int u   = (unsigned int)(key >> 32);
    const unsigned int idx = ~(unsigned int)(key & 0xFFFFFFFFull);
    float b0 = 0.f, b1v = 0.f, b2v = 0.f, b3 = 0.f, sv = 0.f;
    if (key != 0ull && idx < NBOX) {
      sv = fdec(u);
      const float4 bb = *(const float4*)(bp + (size_t)idx * 4);
      b0 = bb.x; b1v = bb.y; b2v = bb.z; b3 = bb.w;
    }
    ry1[m] = b0; rx1[m] = b1v; ry2[m] = b2v; rx2[m] = b3;
    rar[m] = (b2v - b0) * (b3 - b1v);
    rsc[m] = sv;
  }

  // ================= Phase E: greedy NMS, 32-pivot chunks (wave32) ==========
#pragma unroll 1
  for (int c = 0; c < NT / 32; ++c) {
    const int base = c * 32;
    // owners (threads 8c..8c+7) export their 4 boxes + keep nibble
    if ((tid >> 3) == c) {
      const int off = (tid & 7) * 4;
      s_nib[tid & 7] = keepm & 0xFu;
#pragma unroll
      for (int m = 0; m < 4; ++m) {
        const int sl = off + m;
        s_stage[sl]           = ry1[m];
        s_stage[32 + sl]      = rx1[m];
        s_stage[64 + sl]      = ry2[m];
        s_stage[96 + sl]      = rx2[m];
        s_stage[128 + sl]     = rar[m];
      }
    }
    __syncthreads();
    // serial greedy pass over the 32 pivots inside a single wave (no barriers)
    if (tid < 32) {
      const float py1 = s_stage[lane],       px1 = s_stage[32 + lane];
      const float py2 = s_stage[64 + lane],  px2 = s_stage[96 + lane];
      const float par = s_stage[128 + lane];
      const unsigned int aliveBit = (s_nib[lane >> 2] >> (lane & 3)) & 1u;
      unsigned int mask = (unsigned int)__ballot(aliveBit != 0u);
      for (int p = 0; p < 32; ++p) {
        if ((mask >> p) & 1u) {
          const float qy1 = __shfl(py1, p), qx1 = __shfl(px1, p);
          const float qy2 = __shfl(py2, p), qx2 = __shfl(px2, p);
          const float qar = __shfl(par, p);
          const float iy1 = fmaxf(qy1, py1), ix1 = fmaxf(qx1, px1);
          const float iy2 = fminf(qy2, py2), ix2 = fminf(qx2, px2);
          const float inter = fmaxf(iy2 - iy1, 0.0f) * fmaxf(ix2 - ix1, 0.0f);
          const float iou = inter / (qar + par - inter + 1e-9f);
          const bool sup = (((mask >> lane) & 1u) != 0u) && (lane > p) && (iou > IOU_CUT);
          mask &= ~(unsigned int)__ballot(sup);
        }
      }
      if (lane == 0) s_surv = mask;
    }
    __syncthreads();
    const unsigned int surv = s_surv;
    if ((tid >> 3) == c) keepm = (surv >> ((tid & 7) * 4)) & 0xFu;  // sync owners
    // tail suppression: every thread checks its boxes beyond this chunk
#pragma unroll
    for (int m = 0; m < 4; ++m) {
      const int slot = tid * 4 + m;
      if (slot >= base + 32 && ((keepm >> m) & 1u)) {
        unsigned int s = surv;
        while (s) {
          const int p = __ffs(s) - 1; s &= s - 1;
          const float qy1 = s_stage[p],       qx1 = s_stage[32 + p];
          const float qy2 = s_stage[64 + p],  qx2 = s_stage[96 + p];
          const float qar = s_stage[128 + p];
          const float iy1 = fmaxf(qy1, ry1[m]), ix1 = fmaxf(qx1, rx1[m]);
          const float iy2 = fminf(qy2, ry2[m]), ix2 = fminf(qx2, rx2[m]);
          const float inter = fmaxf(iy2 - iy1, 0.0f) * fmaxf(ix2 - ix1, 0.0f);
          const float iou = inter / (qar + rar[m] - inter + 1e-9f);
          if (iou > IOU_CUT) { keepm &= ~(1u << m); break; }
        }
      }
    }
    __syncthreads();   // protect s_stage before next chunk's export
  }

  // ============ Phase F: compaction rank via WMMA prefix scan ==============
  const unsigned int cnt = __popc(keepm & 0xFu);
  s_scan[tid] = (float)cnt;
  __syncthreads();
  const float incl = wave_scan32_wmma(lane, &s_scan[wave * 32]);  // v_wmma x2
  const float wsum = __shfl(incl, 31);
  if (lane == 0) s_wsum[wave] = wsum;
  __syncthreads();
  if (wave == 0) {                                                // v_wmma x2
    const float winc = wave_scan32_wmma(lane, s_wsum);
    s_wsum[32 + lane] = winc - s_wsum[lane];                      // exclusive
  }
  __syncthreads();

  int r = (int)(s_wsum[32 + wave] + incl - (float)cnt);           // exact ints
  float* ob = outB + (size_t)b * POST * 4;
  float* os = outS + (size_t)b * POST;
#pragma unroll
  for (int m = 0; m < 4; ++m) {
    if ((keepm >> m) & 1u) {
      if (r < POST) {
        ob[r * 4 + 0] = fminf(fmaxf(ry1[m], 0.0f), 1.0f);
        ob[r * 4 + 1] = fminf(fmaxf(rx1[m], 0.0f), 1.0f);
        ob[r * 4 + 2] = fminf(fmaxf(ry2[m], 0.0f), 1.0f);
        ob[r * 4 + 3] = fminf(fmaxf(rx2[m], 0.0f), 1.0f);
        os[r] = rsc[m];
      }
      ++r;
    }
  }
}

extern "C" void kernel_launch(void* const* d_in, const int* in_sizes, int n_in,
                              void* d_out, int out_size, void* d_ws, size_t ws_size,
                              hipStream_t stream) {
  (void)in_sizes; (void)n_in; (void)out_size; (void)d_ws; (void)ws_size;
  const float* boxes  = (const float*)d_in[0];   // [16,200000,4] f32
  const float* scores = (const float*)d_in[1];   // [16,200000,1] f32
  float* outB = (float*)d_out;                   // [16,2000,4]
  float* outS = outB + (size_t)BATCH * POST * 4; // [16,2000]
  proposal_nms_kernel<<<dim3(BATCH), dim3(NTHREADS), 0, stream>>>(boxes, scores, outB, outS);
}
